// MultiHeadAttention_41936060678770
// MI455X (gfx1250) — compile-verified, matrix operand order checked by reading, not run on hardware
//
#include <hip/hip_runtime.h>
#include <hip/hip_bf16.h>

// ---------------------------------------------------------------------------
// LinBERT linear-attention block for MI455X (gfx1250, wave32, WMMA + TDM).
// GEMMs: v_wmma_f32_16x16x32_f16 ; tile staging: tensor_load_to_lds (TDM)
// with double-buffered (ping-pong) LDS pipeline in the big GEMMs.
// ---------------------------------------------------------------------------

#ifndef __has_builtin
#define __has_builtin(x) 0
#endif

#if defined(__HIP_DEVICE_COMPILE__) && \
    __has_builtin(__builtin_amdgcn_tensor_load_to_lds) && \
    __has_builtin(__builtin_amdgcn_s_wait_tensorcnt)
#define USE_TDM 1
#else
#define USE_TDM 0
#endif

namespace {

constexpr int HS    = 1024;           // hidden size
constexpr int NH    = 16;             // heads
constexpr int PS    = 64;             // head dim
constexpr int BATCH = 4;
constexpr int SEQ   = 4096;
constexpr int MTOT  = BATCH * SEQ;    // 16384 rows

typedef __attribute__((ext_vector_type(16))) _Float16 v16h;
typedef __attribute__((ext_vector_type(8)))  _Float16 v8h;
typedef __attribute__((ext_vector_type(8)))  float    v8f;
typedef __attribute__((ext_vector_type(4)))  unsigned v4u;
typedef __attribute__((ext_vector_type(8)))  int      v8i;
typedef __attribute__((ext_vector_type(4)))  int      v4i;

#if USE_TDM
// TDM 2D f16 tile load: global (row stride strideElems) -> LDS, with LDS row
// padding.  padInterval/padAmount are the encoded D# codes:
//   row bytes = 2^(padInterval+3), pad bytes = (padAmount+1)*4.
// D# per CDNA5 ISA (count=1, type=2=image,
// data_size code 1 (2 bytes), pad_enable, 2D tile).
__device__ __forceinline__ void tdm_load_tile_f16(
    void* ldsPtr, const _Float16* gptr, unsigned tileW, unsigned tileH,
    unsigned strideElems, unsigned padInterval, unsigned padAmount) {
  const unsigned lds = (unsigned)(uintptr_t)ldsPtr;   // low 32b = LDS offset
  const unsigned long long ga = (unsigned long long)(uintptr_t)gptr;
  v4u g0;
  g0[0] = 1u;                                        // count = 1 valid D#
  g0[1] = lds;                                       // lds_addr
  g0[2] = (unsigned)(ga & 0xffffffffull);            // global_addr[31:0]
  g0[3] = (unsigned)((ga >> 32) & 0x01ffffffull)     // global_addr[56:32]
          | (2u << 30);                              // type = 2 (image)
  v8i g1;
  g1[0] = (int)((1u << 16)                           // data_size = 2 bytes
                | (1u << 20)                         // pad_enable
                | (padInterval << 22) | (padAmount << 25));
  g1[1] = (int)((strideElems & 0xffffu) << 16);      // tensor_dim0 lo16
  g1[2] = (int)((strideElems >> 16) | ((tileH & 0xffffu) << 16)); // dim0 hi / dim1 lo
  g1[3] = (int)((tileH >> 16) | (tileW << 16));      // dim1 hi / tile_dim0
  g1[4] = (int)(tileH & 0xffffu);                    // tile_dim1 (tile_dim2 = 0)
  g1[5] = (int)strideElems;                          // tensor_dim0_stride lo32
  g1[6] = 0;                                         // stride hi / dim1_stride
  g1[7] = 0;
  const v4i z4 = {0, 0, 0, 0};
  const v8i z8 = {0, 0, 0, 0, 0, 0, 0, 0};
  __builtin_amdgcn_tensor_load_to_lds(g0, g1, z4, z4, z8, 0);
}
#define TDM_WAIT(n) __builtin_amdgcn_s_wait_tensorcnt(n)
#else
#define TDM_WAIT(n) ((void)0)
#endif

// Load a 16x32 f16 WMMA A/B fragment from an LDS tile stored row-major.
// ISA layout: lane = row%16 ; halves 0..7 -> K = 8*(lane/16)+h,
//             halves 8..15 -> K = 16 + 8*(lane/16) + (h-8).
__device__ __forceinline__ v16h load_frag(const _Float16* __restrict__ base,
                                          int strideHalves, int lane) {
  const int m  = lane & 15;
  const int kb = (lane >> 4) << 3;
  const _Float16* p = base + m * strideHalves + kb;
  v8h lo = *(const v8h*)(p);
  v8h hi = *(const v8h*)(p + 16);
  return __builtin_shufflevector(lo, hi, 0, 1, 2, 3, 4, 5, 6, 7,
                                 8, 9, 10, 11, 12, 13, 14, 15);
}

// ---------------------------------------------------------------------------
// Utility kernels
// ---------------------------------------------------------------------------

__global__ __launch_bounds__(256) void zero_kernel(float* p, int n) {
  int i = blockIdx.x * 256 + threadIdx.x;
  if (i < n) p[i] = 0.0f;
}

// W[k][n] (f32) -> Wt[n][k] (f16)   (1024x1024)
__global__ __launch_bounds__(256) void wconv_kernel(const float* __restrict__ w,
                                                    _Float16* __restrict__ wt) {
  int idx = blockIdx.x * 256 + threadIdx.x;
  int n = idx >> 10, k = idx & 1023;
  wt[idx] = (_Float16)w[k * HS + n];
}

// kv (f32) -> f16 copy so attn can TDM-load its B tiles directly.
__global__ __launch_bounds__(256) void cvt_kernel(const float* __restrict__ src,
                                                  _Float16* __restrict__ dst, int n) {
  int i = blockIdx.x * 256 + threadIdx.x;
  if (i < n) dst[i] = (_Float16)src[i];
}

// Per-row LayerNorm (1024 wide), output f16.  One block per row, 256 threads.
__global__ __launch_bounds__(256) void ln_kernel(const float* __restrict__ x,
                                                 const float* __restrict__ g,
                                                 const float* __restrict__ bb,
                                                 _Float16* __restrict__ y) {
  __shared__ float red[256];
  const int row = blockIdx.x;
  const int tid = threadIdx.x;
  const float* xr = x + (size_t)row * HS;
  float4 v = *(const float4*)(xr + tid * 4);
  red[tid] = v.x + v.y + v.z + v.w;
  __syncthreads();
  for (int o = 128; o > 0; o >>= 1) {
    if (tid < o) red[tid] += red[tid + o];
    __syncthreads();
  }
  const float mu = red[0] * (1.0f / HS);
  __syncthreads();
  const float dx = v.x - mu, dy = v.y - mu, dz = v.z - mu, dw = v.w - mu;
  red[tid] = dx * dx + dy * dy + dz * dz + dw * dw;
  __syncthreads();
  for (int o = 128; o > 0; o >>= 1) {
    if (tid < o) red[tid] += red[tid + o];
    __syncthreads();
  }
  const float rs = rsqrtf(red[0] * (1.0f / HS) + 1e-5f);
  const int c = tid * 4;
  _Float16* yr = y + (size_t)row * HS + c;
  yr[0] = (_Float16)(dx * rs * g[c + 0] + bb[c + 0]);
  yr[1] = (_Float16)(dy * rs * g[c + 1] + bb[c + 1]);
  yr[2] = (_Float16)(dz * rs * g[c + 2] + bb[c + 2]);
  yr[3] = (_Float16)(dw * rs * g[c + 3] + bb[c + 3]);
}

// ---------------------------------------------------------------------------
// Main GEMM: C[M=16384, N=1024] = A(f16, MxK) * Wt(f16, NxK)^T + bias
// Block tile 128x128, 256 threads = 8 waves (4x2), wave tile 32x64.
// Double-buffered TDM pipeline: issue tile i+1, s_wait_tensorcnt 1 for tile i.
// EPI: 0 = elu+1 -> f16 ; 1 = (elu+1)*mask -> f16 ; 2 = plain -> f16 ;
//      3 = + residual -> f32 (final output)
// ---------------------------------------------------------------------------
template <int EPI>
__global__ __launch_bounds__(256)
void gemm_epi(const _Float16* __restrict__ A, const _Float16* __restrict__ Bt,
              const float* __restrict__ bias, const float* __restrict__ mask,
              const float* __restrict__ residual,
              _Float16* __restrict__ outH, float* __restrict__ outF) {
  constexpr int KT  = 32;
  constexpr int LDT = KT + 8;  // 40 halves -> 80B rows (64B data + 16B pad)
  __shared__ alignas(16) _Float16 As[2][128 * LDT];
  __shared__ alignas(16) _Float16 Bs[2][128 * LDT];

  const int tid  = threadIdx.x;
  const int lane = tid & 31;
  const int w    = tid >> 5;   // 0..7
  const int wm   = w & 3;      // 4 M groups of 32 rows
  const int wn   = w >> 2;     // 2 N groups of 64 cols
  const int rowBase = blockIdx.x * 128;
  const int colBase = blockIdx.y * 128;

  const _Float16* aBase = A  + (size_t)rowBase * HS;
  const _Float16* bBase = Bt + (size_t)colBase * HS;

  v8f acc[2][4] = {};
  constexpr int NIT = HS / KT;   // 32

#if USE_TDM
  // 128x32-half tiles, row 64B (interval code 3), pad 16B (amount code 3)
  if (w == 0) tdm_load_tile_f16(As[0], aBase, KT, 128, HS, 3, 3);
  if (w == 1) tdm_load_tile_f16(Bs[0], bBase, KT, 128, HS, 3, 3);
#endif

  for (int it = 0; it < NIT; ++it) {
    const int cur = it & 1;
#if USE_TDM
    if (it + 1 < NIT) {   // prefetch next K tile into the other buffer
      if (w == 0) tdm_load_tile_f16(As[cur ^ 1], aBase + (it + 1) * KT, KT, 128, HS, 3, 3);
      if (w == 1) tdm_load_tile_f16(Bs[cur ^ 1], bBase + (it + 1) * KT, KT, 128, HS, 3, 3);
      TDM_WAIT(1);        // wait for tile `it` only (per-wave, in-order)
    } else {
      TDM_WAIT(0);
    }
#else
    const int k0 = it * KT;
    const int lr = tid >> 1;        // 0..127
    const int lc = (tid & 1) << 4;  // 0 / 16 halves
    const _Float16* gp = aBase + (size_t)lr * HS + k0 + lc;
    const _Float16* bp = bBase + (size_t)lr * HS + k0 + lc;
    *(v8h*)&As[cur][lr * LDT + lc]     = *(const v8h*)gp;
    *(v8h*)&As[cur][lr * LDT + lc + 8] = *(const v8h*)(gp + 8);
    *(v8h*)&Bs[cur][lr * LDT + lc]     = *(const v8h*)bp;
    *(v8h*)&Bs[cur][lr * LDT + lc + 8] = *(const v8h*)(bp + 8);
#endif
    __syncthreads();

    v16h af[2], bf[4];
#pragma unroll
    for (int mi = 0; mi < 2; ++mi)
      af[mi] = load_frag(&As[cur][(wm * 32 + mi * 16) * LDT], LDT, lane);
#pragma unroll
    for (int ni = 0; ni < 4; ++ni)
      bf[ni] = load_frag(&Bs[cur][(wn * 64 + ni * 16) * LDT], LDT, lane);
#pragma unroll
    for (int mi = 0; mi < 2; ++mi)
#pragma unroll
      for (int ni = 0; ni < 4; ++ni)
        acc[mi][ni] = __builtin_amdgcn_wmma_f32_16x16x32_f16(
            false, af[mi], false, bf[ni], (short)0, acc[mi][ni], false, false);
    __syncthreads();
  }

  // Epilogue.  C/D layout: VGPR r -> M = r + 8*(lane/16), N = lane%16.
  const int rl = lane >> 4;
  const int cl = lane & 15;
#pragma unroll
  for (int mi = 0; mi < 2; ++mi) {
#pragma unroll
    for (int ni = 0; ni < 4; ++ni) {
#pragma unroll
      for (int r = 0; r < 8; ++r) {
        const int row = rowBase + wm * 32 + mi * 16 + r + rl * 8;
        const int col = colBase + wn * 64 + ni * 16 + cl;
        float x = acc[mi][ni][r] + bias[col];
        if (EPI == 0 || EPI == 1) {
          x = (x > 0.0f) ? (x + 1.0f) : __expf(x);   // elu(x)+1
          if (EPI == 1) x *= mask[row];
          outH[(size_t)row * HS + col] = (_Float16)x;
        } else if (EPI == 2) {
          outH[(size_t)row * HS + col] = (_Float16)x;
        } else {
          outF[(size_t)row * HS + col] = x + residual[(size_t)row * HS + col];
        }
      }
    }
  }
}

// ---------------------------------------------------------------------------
// kv[b,n][e][d] = sum_s phi_k[b,s,n,d] * vh[b,s,n,e]   (stored e-major = kv^T)
// z[b,n][d]    = sum_s phi_k[b,s,n,d]
// WMMA version: stage [s][.] tiles, transpose in LDS to [d][s] / [e][s],
// then D(d,e) = PhT(MxK) x VhT^T(KxN) on the matrix pipe.
// grid (64 bn, 16 s-chunks of 256), 256 threads = 8 waves.
// ---------------------------------------------------------------------------
__global__ __launch_bounds__(256)
void kv_kernel(const _Float16* __restrict__ phik, const _Float16* __restrict__ vh,
               float* __restrict__ kvT, float* __restrict__ z) {
  constexpr int LD = 72;  // 144B rows (128B data + 16B pad)
  __shared__ alignas(16) _Float16 Ph [64 * LD];   // [s][d]
  __shared__ alignas(16) _Float16 Vh [64 * LD];   // [s][e]
  __shared__ alignas(16) _Float16 PhT[64 * LD];   // [d][s]
  __shared__ alignas(16) _Float16 VhT[64 * LD];   // [e][s]
  const int tid  = threadIdx.x;
  const int lane = tid & 31;
  const int w    = tid >> 5;
  const int bn   = blockIdx.x;        // b*16+n
  const int b    = bn >> 4, n = bn & 15;
  const int sBase = blockIdx.y * 256;
  const int di  = w & 3;              // d subtile (rows of D)
  const int ei0 = w >> 2;             // e subtiles (cols of D): ei0 and ei0+2
  v8f acc[2] = {};
  float zacc = 0.0f;

  const int tr = tid >> 2;            // 0..63 (transpose source row)
  const int tc = (tid & 3) << 4;      // 0,16,32,48

  for (int sb = 0; sb < 4; ++sb) {
    const int s0 = sBase + sb * 64;
#if USE_TDM
    // 64x64-half tiles, row 128B (interval code 4), pad 16B (amount code 3)
    if (w == 0)
      tdm_load_tile_f16(Ph, phik + (size_t)(b * SEQ + s0) * HS + n * PS,
                        PS, 64, HS, 4, 3);
    if (w == 1)
      tdm_load_tile_f16(Vh, vh + (size_t)(b * SEQ + s0) * HS + n * PS,
                        PS, 64, HS, 4, 3);
    TDM_WAIT(0);
#else
    const _Float16* pp = phik + (size_t)(b * SEQ + s0 + tr) * HS + n * PS + tc;
    const _Float16* vp = vh   + (size_t)(b * SEQ + s0 + tr) * HS + n * PS + tc;
    *(v8h*)&Ph[tr * LD + tc]     = *(const v8h*)pp;
    *(v8h*)&Ph[tr * LD + tc + 8] = *(const v8h*)(pp + 8);
    *(v8h*)&Vh[tr * LD + tc]     = *(const v8h*)vp;
    *(v8h*)&Vh[tr * LD + tc + 8] = *(const v8h*)(vp + 8);
#endif
    __syncthreads();
    // LDS transpose: 16 contiguous reads, 16 strided writes per thread.
#pragma unroll
    for (int i = 0; i < 16; ++i) {
      PhT[(tc + i) * LD + tr] = Ph[tr * LD + tc + i];
      VhT[(tc + i) * LD + tr] = Vh[tr * LD + tc + i];
    }
    __syncthreads();
#pragma unroll
    for (int kk = 0; kk < 64; kk += 32) {
      const v16h a  = load_frag(&PhT[(di * 16) * LD + kk], LD, lane);
      const v16h b0 = load_frag(&VhT[(ei0 * 16) * LD + kk], LD, lane);
      const v16h b1 = load_frag(&VhT[((ei0 + 2) * 16) * LD + kk], LD, lane);
      acc[0] = __builtin_amdgcn_wmma_f32_16x16x32_f16(
          false, a, false, b0, (short)0, acc[0], false, false);
      acc[1] = __builtin_amdgcn_wmma_f32_16x16x32_f16(
          false, a, false, b1, (short)0, acc[1], false, false);
    }
    if (tid < 64) {  // z: row sums of PhT (contiguous)
      float s = 0.0f;
      for (int ss = 0; ss < 64; ++ss) s += (float)PhT[tid * LD + ss];
      zacc += s;
    }
    __syncthreads();
  }
  // D layout: VGPR r -> d = di*16 + r + 8*(lane/16) ; e = ei*16 + lane%16.
  const int dd = di * 16 + (lane >> 4) * 8;
  const int ee = lane & 15;
#pragma unroll
  for (int j = 0; j < 2; ++j) {
    const int e = (ei0 + 2 * j) * 16 + ee;
#pragma unroll
    for (int r = 0; r < 8; ++r)
      atomicAdd(&kvT[(size_t)bn * (PS * PS) + e * PS + dd + r], acc[j][r]);
  }
  if (tid < 64) atomicAdd(&z[bn * PS + tid], zacc);
}

// ---------------------------------------------------------------------------
// attn[b,s,n,:] = (phi_q[b,s,n,:] @ kv[b,n]) / (phi_q . z + eps)  -> f16
// grid (64 bn, 32 s-tiles of 128), 256 threads = 8 waves, wave tile 16x64.
// One reciprocal per row (not per element); epilogue multiplies.
// ---------------------------------------------------------------------------
__global__ __launch_bounds__(256)
void attn_kernel(const _Float16* __restrict__ phiq, const _Float16* __restrict__ kvh,
                 const float* __restrict__ z, _Float16* __restrict__ attn) {
  constexpr int LD = 72;
  __shared__ alignas(16) _Float16 As[128 * LD];
  __shared__ alignas(16) _Float16 Bs[64 * LD];
  __shared__ float zs[64];
  __shared__ float rden[128];
  const int tid  = threadIdx.x;
  const int lane = tid & 31;
  const int w    = tid >> 5;
  const int bn   = blockIdx.x;
  const int b    = bn >> 4, n = bn & 15;
  const int sBase = blockIdx.y * 128;

#if USE_TDM
  if (w == 0)  // 128x64-half tile, row 128B (code 4), pad 16B (code 3)
    tdm_load_tile_f16(As, phiq + (size_t)(b * SEQ + sBase) * HS + n * PS,
                      PS, 128, HS, 4, 3);
  if (w == 1)  // kv^T f16 [e][d], contiguous 64-wide rows
    tdm_load_tile_f16(Bs, kvh + (size_t)bn * (PS * PS), PS, 64, PS, 4, 3);
#else
  {
    const int r = tid >> 1;
    const int c = (tid & 1) << 5;
    const _Float16* ap = phiq + (size_t)(b * SEQ + sBase + r) * HS + n * PS + c;
    *(v8h*)&As[r * LD + c]      = *(const v8h*)ap;
    *(v8h*)&As[r * LD + c + 8]  = *(const v8h*)(ap + 8);
    *(v8h*)&As[r * LD + c + 16] = *(const v8h*)(ap + 16);
    *(v8h*)&As[r * LD + c + 24] = *(const v8h*)(ap + 24);
  }
  for (int i = tid; i < PS * PS; i += 256) {
    const int ee = i >> 6, dd = i & 63;
    Bs[ee * LD + dd] = kvh[(size_t)bn * (PS * PS) + i];
  }
#endif
  if (tid < 64) zs[tid] = z[bn * PS + tid];
  TDM_WAIT(0);
  __syncthreads();

  if (tid < 128) {
    float ds = 0.0f;
    for (int kk = 0; kk < 64; ++kk) ds += (float)As[tid * LD + kk] * zs[kk];
    rden[tid] = 1.0f / (ds + 1e-6f);       // one divide per row
  }
  __syncthreads();

  v8f acc[4] = {};
#pragma unroll
  for (int kk = 0; kk < 64; kk += 32) {
    const v16h a = load_frag(&As[(w * 16) * LD + kk], LD, lane);
#pragma unroll
    for (int ni = 0; ni < 4; ++ni) {
      const v16h bb = load_frag(&Bs[(ni * 16) * LD + kk], LD, lane);
      acc[ni] = __builtin_amdgcn_wmma_f32_16x16x32_f16(
          false, a, false, bb, (short)0, acc[ni], false, false);
    }
  }
  const int rl = lane >> 4, cl = lane & 15;
#pragma unroll
  for (int ni = 0; ni < 4; ++ni)
#pragma unroll
    for (int r = 0; r < 8; ++r) {
      const int rowL = w * 16 + r + rl * 8;
      const float v  = acc[ni][r] * rden[rowL];
      const int row  = b * SEQ + sBase + rowL;
      const int col  = n * PS + ni * 16 + cl;
      attn[(size_t)row * HS + col] = (_Float16)v;
    }
}

}  // namespace

// ---------------------------------------------------------------------------
// Host-side orchestration (graph-capture safe: only kernel launches on stream)
// ---------------------------------------------------------------------------
extern "C" void kernel_launch(void* const* d_in, const int* in_sizes, int n_in,
                              void* d_out, int out_size, void* d_ws, size_t ws_size,
                              hipStream_t stream) {
  const float* q      = (const float*)d_in[0];
  const float* k      = (const float*)d_in[1];
  const float* v      = (const float*)d_in[2];
  const float* ln_q_g = (const float*)d_in[3];
  const float* ln_q_b = (const float*)d_in[4];
  const float* wq     = (const float*)d_in[5];
  const float* bq     = (const float*)d_in[6];
  const float* ln_k_g = (const float*)d_in[7];
  const float* ln_k_b = (const float*)d_in[8];
  const float* wk     = (const float*)d_in[9];
  const float* bk     = (const float*)d_in[10];
  const float* ln_v_g = (const float*)d_in[11];
  const float* ln_v_b = (const float*)d_in[12];
  const float* wv     = (const float*)d_in[13];
  const float* bv     = (const float*)d_in[14];
  const float* fc_w   = (const float*)d_in[15];
  const float* fc_b   = (const float*)d_in[16];
  const float* mask   = (const float*)d_in[17];
  float* out = (float*)d_out;

  const size_t BIG = (size_t)MTOT * HS;      // 16M elements
  const int KVN = BATCH * NH * PS * PS;      // 262144
  char* p = (char*)d_ws;
  _Float16* lnq  = (_Float16*)p;                 p += BIG * 2;
  _Float16* lnk  = (_Float16*)p;                 p += BIG * 2;
  _Float16* lnv  = (_Float16*)p;                 p += BIG * 2;
  _Float16* phiq = (_Float16*)p;                 p += BIG * 2;
  _Float16* phik = (_Float16*)p;                 p += BIG * 2;
  _Float16* vhb  = (_Float16*)p;                 p += BIG * 2;
  _Float16* wqt  = (_Float16*)p;                 p += (size_t)HS * HS * 2;
  _Float16* wkt  = (_Float16*)p;                 p += (size_t)HS * HS * 2;
  _Float16* wvt  = (_Float16*)p;                 p += (size_t)HS * HS * 2;
  _Float16* fct  = (_Float16*)p;                 p += (size_t)HS * HS * 2;
  float*    kvT  = (float*)p;                    p += (size_t)KVN * 4;
  float*    zbuf = (float*)p;                    p += (size_t)BATCH * NH * PS * 4;
  _Float16* kvh  = (_Float16*)p;                 p += (size_t)KVN * 2;
  _Float16* attn = lnq;  // lnq is dead by the time attn is written

  const int nzero = KVN + BATCH * NH * PS;   // kvT + z contiguous
  zero_kernel<<<(nzero + 255) / 256, 256, 0, stream>>>(kvT, nzero);

  wconv_kernel<<<HS * HS / 256, 256, 0, stream>>>(wq,   wqt);
  wconv_kernel<<<HS * HS / 256, 256, 0, stream>>>(wk,   wkt);
  wconv_kernel<<<HS * HS / 256, 256, 0, stream>>>(wv,   wvt);
  wconv_kernel<<<HS * HS / 256, 256, 0, stream>>>(fc_w, fct);

  ln_kernel<<<MTOT, 256, 0, stream>>>(q, ln_q_g, ln_q_b, lnq);
  ln_kernel<<<MTOT, 256, 0, stream>>>(k, ln_k_g, ln_k_b, lnk);
  ln_kernel<<<MTOT, 256, 0, stream>>>(v, ln_v_g, ln_v_b, lnv);

  const dim3 gg(MTOT / 128, HS / 128);
  gemm_epi<0><<<gg, 256, 0, stream>>>(lnq, wqt, bq, nullptr, nullptr, phiq, nullptr);
  gemm_epi<1><<<gg, 256, 0, stream>>>(lnk, wkt, bk, mask,    nullptr, phik, nullptr);
  gemm_epi<2><<<gg, 256, 0, stream>>>(lnv, wvt, bv, nullptr, nullptr, vhb,  nullptr);

  kv_kernel<<<dim3(BATCH * NH, SEQ / 256), 256, 0, stream>>>(phik, vhb, kvT, zbuf);
  cvt_kernel<<<(KVN + 255) / 256, 256, 0, stream>>>(kvT, kvh, KVN);
  attn_kernel<<<dim3(BATCH * NH, SEQ / 128), 256, 0, stream>>>(phiq, kvh, zbuf, attn);

  gemm_epi<3><<<gg, 256, 0, stream>>>(attn, fct, fc_b, nullptr, q, nullptr, out);
}
